// PGD_37847251812886
// MI455X (gfx1250) — matrix operand order repaired and checked by reading, not compiled, not run on hardware
//
#include <hip/hip_runtime.h>
#include <hip/hip_bf16.h>

// Problem constants (match reference)
#define B_  2
#define S_  1024
#define D_  768
#define H_  12
#define V_  16384
#define L_  4
#define SP1 1040   // (S_+1)=1025 padded up to a multiple of 16

typedef __attribute__((ext_vector_type(16))) __bf16 v16bf;
typedef __attribute__((ext_vector_type(8)))  __bf16 v8bf;
typedef __attribute__((ext_vector_type(8)))  float  v8f;

// ---------------------------------------------------------------------------
// WMMA helpers (CDNA5 wave32, V_WMMA_F32_16X16X32_BF16)
// A (16x32 MxK): lane m=lane&15; lane<16 holds K={0..7,16..23}, lane>=16
// holds K={8..15,24..31}.  B (32x16 KxN): lane n=lane&15 holds column n, same K split as A.
// C/D (16x16 f32): VGPR r, lanes 0-15 -> (M=r, N=lane), lanes 16-31 -> M=r+8.
// ---------------------------------------------------------------------------
__device__ __forceinline__ v8f wmma_bf16(v16bf a, v16bf b, v8f c) {
  return __builtin_amdgcn_wmma_f32_16x16x32_bf16(
      false, a, false, b, (short)0, c, false, false);
}

__device__ __forceinline__ v16bf load_a_frag(const __bf16* A0, int lda) {
  int lane = threadIdx.x & 31;
  int m    = lane & 15;
  int ko   = (lane & 16) ? 8 : 0;
  const __bf16* p = A0 + (size_t)m * lda + ko;
  v8bf lo = *(const v8bf*)p;         // K = ko .. ko+7
  v8bf hi = *(const v8bf*)(p + 16);  // K = ko+16 .. ko+23
  v16bf r;
#pragma unroll
  for (int i = 0; i < 8; ++i) { r[i] = lo[i]; r[8 + i] = hi[i]; }
  return r;
}

__device__ __forceinline__ v16bf load_b_frag(const __bf16* B0, int ldb) {
  int lane = threadIdx.x & 31;
  int n    = lane & 15;
  int ko   = (lane & 16) ? 8 : 0;
  const __bf16* p = B0 + n;  // column n=lane&15, K = ko..ko+7, ko+16..ko+23
  v16bf r;
#pragma unroll
  for (int i = 0; i < 8; ++i) {
    r[i]     = p[(size_t)(ko + i) * ldb];
    r[8 + i] = p[(size_t)(ko + 16 + i) * ldb];
  }
  return r;
}

__device__ __forceinline__ void store_c_f32(float* C0, int ldc, v8f acc) {
  int lane = threadIdx.x & 31;
  int n = lane & 15, mb = (lane & 16) ? 8 : 0;
#pragma unroll
  for (int r = 0; r < 8; ++r) C0[(size_t)(mb + r) * ldc + n] = acc[r];
}
__device__ __forceinline__ void store_c_bf16(__bf16* C0, int ldc, v8f acc) {
  int lane = threadIdx.x & 31;
  int n = lane & 15, mb = (lane & 16) ? 8 : 0;
#pragma unroll
  for (int r = 0; r < 8; ++r) C0[(size_t)(mb + r) * ldc + n] = (__bf16)acc[r];
}
__device__ __forceinline__ void store_c_bf16_T(__bf16* Ct0, int ldct, v8f acc) {
  int lane = threadIdx.x & 31;
  int n = lane & 15, mb = (lane & 16) ? 8 : 0;
#pragma unroll
  for (int r = 0; r < 8; ++r) Ct0[(size_t)n * ldct + mb + r] = (__bf16)acc[r];
}

// ---------------------------------------------------------------------------
// CDNA5 async global->LDS copy (ASYNCcnt path, CDNA5 ISA).
// GV mode: VDST = LDS byte-address VGPR, VADDR = 64-bit global address pair.
// Generic LDS pointers carry the LDS byte offset in addr[31:0] (ISA 10.2).
// ---------------------------------------------------------------------------
__device__ __forceinline__ void async_copy16(void* lds_ptr, const void* gptr) {
  unsigned lds_off = (unsigned)(uintptr_t)lds_ptr;
  unsigned long long ga = (unsigned long long)(uintptr_t)gptr;
  asm volatile("global_load_async_to_lds_b128 %0, %1, off"
               :: "v"(lds_off), "v"(ga) : "memory");
}
__device__ __forceinline__ void wait_async0() {
  asm volatile("s_wait_asynccnt 0x0" ::: "memory");
}
__device__ __forceinline__ void wait_async1() {
  asm volatile("s_wait_asynccnt 0x1" ::: "memory");
}

// ---------------------------------------------------------------------------
// Simple bf16 GEMM (pre-loop, one-time QK work): one 16x16 tile per wave.
// mode: 0 = f32 out, 1 = bf16 out, 2 = bf16 transposed out.
// ---------------------------------------------------------------------------
__global__ __launch_bounds__(256) void k_gemm(
    const __bf16* __restrict__ A, int lda,
    const __bf16* __restrict__ B, int ldb,
    void* __restrict__ C, int ldc, int N, int K, int mode)
{
  int wave = threadIdx.x >> 5;
  int m0 = blockIdx.x * 16;
  int n0 = (blockIdx.y * 8 + wave) * 16;
  if (n0 >= N) return;                     // wave-uniform guard
  v8f acc = {};
  const __bf16* Arow = A + (size_t)m0 * lda;
  for (int k = 0; k < K; k += 32) {
    v16bf a = load_a_frag(Arow + k, lda);
    v16bf b = load_b_frag(B + (size_t)k * ldb + n0, ldb);
    acc = wmma_bf16(a, b, acc);
  }
  if (mode == 0)      store_c_f32 ((float*)C  + (size_t)m0 * ldc + n0, ldc, acc);
  else if (mode == 1) store_c_bf16((__bf16*)C + (size_t)m0 * ldc + n0, ldc, acc);
  else                store_c_bf16_T((__bf16*)C + (size_t)n0 * ldc + m0, ldc, acc);
}

// ---------------------------------------------------------------------------
// Blocked bf16 GEMM body: block tile 32x256, per-wave 16x64 (4 accumulators,
// A-fragment reused x4).  A is staged in LDS with double-buffered async
// copies (chunk K=64); B streams from L2.  Requires M%32==0, N%256==0,
// K%64==0.  LDS rows padded to 72 elements (144 B) for bank spread.
// ---------------------------------------------------------------------------
__device__ __forceinline__ void gemm_big_body(
    const __bf16* __restrict__ A, int lda,
    const __bf16* __restrict__ B, int ldb,
    void* __restrict__ C, int ldc, int K, int mode,
    size_t m0, size_t n0base)
{
  __shared__ __bf16 sA[2][32 * 72];
  int tid = threadIdx.x;
  int wave = tid >> 5;
  int mt = wave >> 2, ng = wave & 3;
  size_t n0 = n0base + (size_t)ng * 64;

  // async A staging: thread t covers row = t/8 (0..31), 16-byte segment t%8
  int arow = tid >> 3, aseg = tid & 7;
  const __bf16* Ag = A + (m0 + arow) * lda + aseg * 8;
  __bf16* Ls0 = &sA[0][arow * 72 + aseg * 8];
  __bf16* Ls1 = &sA[1][arow * 72 + aseg * 8];

  v8f acc[4];
#pragma unroll
  for (int j = 0; j < 4; ++j) { v8f z = {}; acc[j] = z; }

  const int NC = K / 64;
  async_copy16(Ls0, Ag);
  for (int c = 0; c < NC; ++c) {
    int buf = c & 1;
    if (c + 1 < NC) {
      async_copy16(buf ? Ls0 : Ls1, Ag + (size_t)(c + 1) * 64);
      wait_async1();                        // chunk c has landed
    } else {
      wait_async0();
    }
    __syncthreads();
    const __bf16* Bc = B + (size_t)(c * 64) * ldb + n0;
    // warm next chunk's B rows in L2
    if (c + 1 < NC) {
      __builtin_prefetch(Bc + (size_t)64 * ldb + (size_t)(tid & 31) * ldb, 0, 1);
      __builtin_prefetch(Bc + (size_t)96 * ldb + (size_t)(tid & 31) * ldb, 0, 1);
    }
#pragma unroll
    for (int kk = 0; kk < 64; kk += 32) {
      v16bf a = load_a_frag(&sA[buf][(mt * 16) * 72 + kk], 72);
#pragma unroll
      for (int j = 0; j < 4; ++j) {
        v16bf bb = load_b_frag(Bc + (size_t)kk * ldb + j * 16, ldb);
        acc[j] = wmma_bf16(a, bb, acc[j]);
      }
    }
    __syncthreads();
  }

  if (mode == 0) {
    float* Cp = (float*)C + (m0 + (size_t)mt * 16) * ldc + n0;
#pragma unroll
    for (int j = 0; j < 4; ++j) store_c_f32(Cp + j * 16, ldc, acc[j]);
  } else {
    __bf16* Cp = (__bf16*)C + (m0 + (size_t)mt * 16) * ldc + n0;
#pragma unroll
    for (int j = 0; j < 4; ++j) store_c_bf16(Cp + j * 16, ldc, acc[j]);
  }
}

__global__ __launch_bounds__(256) void k_gemm_big(
    const __bf16* __restrict__ A, int lda,
    const __bf16* __restrict__ B, int ldb,
    void* __restrict__ C, int ldc, int K, int mode)
{
  gemm_big_body(A, lda, B, ldb, C, ldc, K, mode,
                (size_t)blockIdx.x * 32, (size_t)blockIdx.y * 256);
}

// Vh[b,h] = diff16[b] @ Wv16[h]  (batched over z = b*H + h)
__global__ __launch_bounds__(256) void k_vh(
    const __bf16* __restrict__ diff16, const __bf16* __restrict__ Wv16,
    __bf16* __restrict__ Vh16)
{
  int z = blockIdx.z, b = z / H_, h = z % H_;
  gemm_big_body(diff16 + (size_t)b * S_ * D_, D_,
                Wv16 + (size_t)h * D_ * D_, D_,
                Vh16 + (size_t)z * S_ * D_, D_, D_, 1,
                (size_t)blockIdx.x * 32, (size_t)blockIdx.y * 256);
}

// ---------------------------------------------------------------------------
// Fused softmax-expectation:  t = clip(f_k @ We^T), E = (exp(t) @ We)/rowsum,
// diff = e - E.  One block per 16 rows of (B*S); 8 waves.  f_k rows staged
// into LDS with async copies; V traversed in 256-wide chunks: phase 1
// per-wave 16x32 logits tile, exp -> LDS; phase 2 all waves accumulate their
// 16x96 slice of exp@We in registers.
// ---------------------------------------------------------------------------
__global__ __launch_bounds__(256) void k_expect(
    const __bf16* __restrict__ fk16, const __bf16* __restrict__ We16,
    const __bf16* __restrict__ WeT16, const float* __restrict__ e32,
    float* __restrict__ diff32, __bf16* __restrict__ diff16)
{
  __shared__ __bf16 sF[16 * D_];    // 24 KB: f_k rows (bf16)
  __shared__ __bf16 sE[16 * 256];   //  8 KB: exp tile (K-chunk of 256)
  __shared__ float  sSum[16];

  int tid  = threadIdx.x;
  int wave = tid >> 5, lane = tid & 31;
  int mbase = (lane & 16) ? 8 : 0;
  int nn    = lane & 15;
  size_t m0 = (size_t)blockIdx.x * 16;

  for (int i = tid; i < 16 * D_ / 8; i += 256)
    async_copy16(sF + (size_t)i * 8, fk16 + m0 * D_ + (size_t)i * 8);
  wait_async0();
  if (tid < 16) sSum[tid] = 0.f;
  __syncthreads();

  v8f eacc[6];
#pragma unroll
  for (int j = 0; j < 6; ++j) { v8f z = {}; eacc[j] = z; }
  float sloc[8] = {0.f, 0.f, 0.f, 0.f, 0.f, 0.f, 0.f, 0.f};

  for (int vc = 0; vc < V_ / 256; ++vc) {
    int vbase = vc * 256 + wave * 32;
    // ---- phase 1: two 16x16 logits tiles, K-reduce over D ----
    v8f t0 = {}, t1 = {};
    for (int k = 0; k < D_; k += 32) {
      v16bf a  = load_a_frag(sF + k, D_);
      v16bf b0 = load_b_frag(WeT16 + (size_t)k * V_ + vbase, V_);
      v16bf b1 = load_b_frag(WeT16 + (size_t)k * V_ + vbase + 16, V_);
      t0 = wmma_bf16(a, b0, t0);
      t1 = wmma_bf16(a, b1, t1);
    }
    __syncthreads();  // previous chunk's phase-2 reads of sE are done
#pragma unroll
    for (int r = 0; r < 8; ++r) {
      float x0 = __expf(fminf(fmaxf(t0[r], -10.f), 10.f));
      float x1 = __expf(fminf(fmaxf(t1[r], -10.f), 10.f));
      sloc[r] += x0 + x1;
      int m = mbase + r;
      sE[m * 256 + wave * 32 + nn]      = (__bf16)x0;
      sE[m * 256 + wave * 32 + 16 + nn] = (__bf16)x1;
    }
    __syncthreads();
    // ---- phase 2: eacc += exp_tile(16x256) @ We[vchunk, :] ----
    for (int ks = 0; ks < 256; ks += 32) {
      v16bf a = load_a_frag(sE + ks, 256);
#pragma unroll
      for (int j = 0; j < 6; ++j) {
        int d0 = (wave * 6 + j) * 16;
        v16bf b = load_b_frag(We16 + (size_t)(vc * 256 + ks) * D_ + d0, D_);
        eacc[j] = wmma_bf16(a, b, eacc[j]);
      }
    }
  }

  // row-sum reduction: 16 lanes per half share the same 8 M-rows
#pragma unroll
  for (int r = 0; r < 8; ++r) {
    float v = sloc[r];
    v += __shfl_xor(v, 1); v += __shfl_xor(v, 2);
    v += __shfl_xor(v, 4); v += __shfl_xor(v, 8);
    if ((lane & 15) == 0) atomicAdd(&sSum[mbase + r], v);
  }
  __syncthreads();

#pragma unroll
  for (int j = 0; j < 6; ++j) {
    int d0 = (wave * 6 + j) * 16;
#pragma unroll
    for (int r = 0; r < 8; ++r) {
      int m = mbase + r;
      float denom = sSum[m] + 1e-8f;
      size_t gi = (m0 + m) * D_ + d0 + nn;
      float dv = e32[gi] - eacc[j][r] / denom;
      diff32[gi] = dv;
      diff16[gi] = (__bf16)dv;
    }
  }
}

// ---------------------------------------------------------------------------
// delta update: f_k += (sum_h (Kscaled[h]@Vh[b,h]) + B_LR*diffsum[b,:]) / S
// (A_LR is pre-folded into Kscaled).  Per-wave 16x64, A-fragment reused x4.
// grid (SP1/16, 2, B); waves cover the 12 N-groups of 64.
// ---------------------------------------------------------------------------
__global__ __launch_bounds__(256) void k_delta(
    const __bf16* __restrict__ K16, const __bf16* __restrict__ Vh16,
    const float* __restrict__ dsum, const float* __restrict__ B_LR,
    float* __restrict__ fk32)
{
  int wave = threadIdx.x >> 5, lane = threadIdx.x & 31;
  int m0 = blockIdx.x * 16;
  int ngrp = blockIdx.y * 8 + wave;
  if (ngrp >= D_ / 64) return;             // wave-uniform guard (12 groups)
  int n0 = ngrp * 64;
  int b  = blockIdx.z;

  v8f acc[4];
#pragma unroll
  for (int j = 0; j < 4; ++j) { v8f z = {}; acc[j] = z; }

  for (int h = 0; h < H_; ++h) {
    const __bf16* Ah = K16 + (size_t)h * SP1 * S_ + (size_t)m0 * S_;
    const __bf16* Bh = Vh16 + ((size_t)(b * H_ + h)) * S_ * D_ + n0;
    for (int k = 0; k < S_; k += 32) {
      v16bf a = load_a_frag(Ah + k, S_);
#pragma unroll
      for (int j = 0; j < 4; ++j) {
        v16bf bb = load_b_frag(Bh + (size_t)k * D_ + j * 16, D_);
        acc[j] = wmma_bf16(a, bb, acc[j]);
      }
    }
  }
  int mb = (lane & 16) ? 8 : 0;
  float blr = B_LR[0];
  const float invN = 1.0f / (float)S_;
#pragma unroll
  for (int j = 0; j < 4; ++j) {
    int n = n0 + j * 16 + (lane & 15);
    float bs = blr * dsum[b * D_ + n];
#pragma unroll
    for (int r = 0; r < 8; ++r) {
      size_t gi = ((size_t)b * SP1 + m0 + mb + r) * D_ + n;
      fk32[gi] += (acc[j][r] + bs) * invN;
    }
  }
}

// ---------------------------------------------------------------------------
// Row softmax over 1025 score columns -> bf16 K scaled by A_LR[h] (first
// 1024 cols kept); pad rows (>=1025) zeroed.  One block/row, launched per h.
// ---------------------------------------------------------------------------
__global__ __launch_bounds__(256) void k_softmax(
    const float* __restrict__ scores, __bf16* __restrict__ K16h,
    const float* __restrict__ alr)
{
  int r = blockIdx.x, tid = threadIdx.x;
  if (r >= S_ + 1) {
    for (int c = tid; c < S_; c += 256) K16h[(size_t)r * S_ + c] = (__bf16)0.f;
    return;
  }
  const float* row = scores + (size_t)r * SP1;
  __shared__ float red[256];
  float mx = -3.4e38f;
  for (int c = tid; c <= S_; c += 256) mx = fmaxf(mx, row[c]);
  red[tid] = mx; __syncthreads();
  for (int o = 128; o > 0; o >>= 1) { if (tid < o) red[tid] = fmaxf(red[tid], red[tid + o]); __syncthreads(); }
  float m = red[0]; __syncthreads();
  float sm = 0.f;
  for (int c = tid; c <= S_; c += 256) sm += __expf(row[c] - m);
  red[tid] = sm; __syncthreads();
  for (int o = 128; o > 0; o >>= 1) { if (tid < o) red[tid] += red[tid + o]; __syncthreads(); }
  float scale = alr[0] / red[0];
  for (int c = tid; c < S_; c += 256)
    K16h[(size_t)r * S_ + c] = (__bf16)(__expf(row[c] - m) * scale);
}

// ----------------------------- utility kernels -----------------------------
__global__ void k_zero_f32(float* p, size_t n) {
  size_t i = (size_t)blockIdx.x * blockDim.x + threadIdx.x;
  if (i < n) p[i] = 0.f;
}
__global__ void k_cast(const float* __restrict__ s, __bf16* __restrict__ d, size_t n) {
  size_t i = (size_t)blockIdx.x * blockDim.x + threadIdx.x;
  if (i < n) d[i] = (__bf16)s[i];
}
__global__ void k_cast_pad_rows(const float* __restrict__ s, __bf16* __restrict__ d,
                                int rows_src, int rows_dst, int cols) {
  size_t i = (size_t)blockIdx.x * blockDim.x + threadIdx.x;
  if (i >= (size_t)rows_dst * cols) return;
  int r = (int)(i / cols);
  d[i] = (r < rows_src) ? (__bf16)s[i] : (__bf16)0.f;
}
__global__ void k_transpose_cast(const float* __restrict__ s /*VxD*/, __bf16* __restrict__ d /*DxV*/) {
  size_t i = (size_t)blockIdx.x * blockDim.x + threadIdx.x;
  if (i >= (size_t)D_ * V_) return;
  int dd = (int)(i / V_), v = (int)(i % V_);
  d[i] = (__bf16)s[(size_t)v * D_ + dd];
}
__global__ void k_gather(const int* __restrict__ idx, const float* __restrict__ We,
                         float* __restrict__ e32) {
  int row = blockIdx.x;                       // 0..B*S-1
  const float* src = We + (size_t)idx[row] * D_;
  float* dst = e32 + (size_t)row * D_;
  for (int d = threadIdx.x; d < D_; d += blockDim.x) dst[d] = src[d];
}
__global__ void k_cast_fk(const float* __restrict__ fk32, __bf16* __restrict__ fk16) {
  size_t i = (size_t)blockIdx.x * blockDim.x + threadIdx.x;
  if (i >= (size_t)B_ * S_ * D_) return;
  int row = (int)(i / D_), d = (int)(i % D_);
  int b = row / S_, s = row % S_;
  fk16[i] = (__bf16)fk32[((size_t)b * SP1 + s) * D_ + d];
}
__global__ void k_diffsum(const float* __restrict__ diff32, float* __restrict__ ds) {
  int i = blockIdx.x * blockDim.x + threadIdx.x;   // 0..B*D-1
  if (i >= B_ * D_) return;
  int b = i / D_, d = i % D_;
  float s = 0.f;
  for (int t = 0; t < S_; ++t) s += diff32[((size_t)b * S_ + t) * D_ + d];
  ds[i] = s;
}

// ---------------------------------------------------------------------------
extern "C" void kernel_launch(void* const* d_in, const int* in_sizes, int n_in,
                              void* d_out, int out_size, void* d_ws, size_t ws_size,
                              hipStream_t stream) {
  (void)in_sizes; (void)n_in; (void)out_size; (void)ws_size;
  const int*   idx  = (const int*)d_in[0];
  const float* W_e  = (const float*)d_in[1];
  const float* W_p  = (const float*)d_in[2];
  const float* W_k  = (const float*)d_in[3];
  const float* W_q  = (const float*)d_in[4];
  const float* W_v  = (const float*)d_in[5];
  const float* A_LR = (const float*)d_in[6];
  const float* B_LR = (const float*)d_in[7];
  float* out = (float*)d_out;

  char* ws = (char*)d_ws;
  size_t off = 0;
  auto alloc = [&](size_t bytes) -> char* {
    char* p = ws + off;
    off += (bytes + 255) & ~(size_t)255;
    return p;
  };
  __bf16* We16   = (__bf16*)alloc((size_t)V_ * D_ * 2);
  __bf16* WeT16  = (__bf16*)alloc((size_t)D_ * V_ * 2);
  __bf16* Wk16   = (__bf16*)alloc((size_t)H_ * D_ * D_ * 2);
  __bf16* Wq16   = (__bf16*)alloc((size_t)H_ * D_ * D_ * 2);
  __bf16* Wv16   = (__bf16*)alloc((size_t)H_ * D_ * D_ * 2);
  __bf16* Wp16   = (__bf16*)alloc((size_t)SP1 * D_ * 2);
  __bf16* xq16   = (__bf16*)alloc((size_t)SP1 * D_ * 2);
  __bf16* xkT16  = (__bf16*)alloc((size_t)D_ * SP1 * 2);
  float*  scores = (float*) alloc((size_t)SP1 * SP1 * 4);
  __bf16* K16    = (__bf16*)alloc((size_t)H_ * SP1 * S_ * 2);
  float*  e32    = (float*) alloc((size_t)B_ * S_ * D_ * 4);
  float*  fk32   = (float*) alloc((size_t)B_ * SP1 * D_ * 4);
  __bf16* fk16   = (__bf16*)alloc((size_t)B_ * S_ * D_ * 2);
  float*  diff32 = (float*) alloc((size_t)B_ * S_ * D_ * 4);
  __bf16* diff16 = (__bf16*)alloc((size_t)B_ * S_ * D_ * 2);
  float*  dsum   = (float*) alloc((size_t)B_ * D_ * 4);
  __bf16* Vh16   = (__bf16*)alloc((size_t)B_ * H_ * S_ * D_ * 2);

  auto g1 = [](size_t n) { return dim3((unsigned)((n + 255) / 256)); };

  // ---- one-time precompute ----
  k_zero_f32<<<g1((size_t)B_ * SP1 * D_), 256, 0, stream>>>(fk32, (size_t)B_ * SP1 * D_);
  k_cast<<<g1((size_t)V_ * D_), 256, 0, stream>>>(W_e, We16, (size_t)V_ * D_);
  k_transpose_cast<<<g1((size_t)D_ * V_), 256, 0, stream>>>(W_e, WeT16);
  k_cast<<<g1((size_t)H_ * D_ * D_), 256, 0, stream>>>(W_k, Wk16, (size_t)H_ * D_ * D_);
  k_cast<<<g1((size_t)H_ * D_ * D_), 256, 0, stream>>>(W_q, Wq16, (size_t)H_ * D_ * D_);
  k_cast<<<g1((size_t)H_ * D_ * D_), 256, 0, stream>>>(W_v, Wv16, (size_t)H_ * D_ * D_);
  k_cast_pad_rows<<<g1((size_t)SP1 * D_), 256, 0, stream>>>(W_p, Wp16, S_ + 1, SP1, D_);
  k_gather<<<dim3(B_ * S_), 256, 0, stream>>>(idx, W_e, e32);

  // ---- attention kernel K (batch-independent), per head ----
  for (int h = 0; h < H_; ++h) {
    k_gemm<<<dim3(SP1 / 16, D_ / 128), 256, 0, stream>>>(
        Wp16, D_, Wq16 + (size_t)h * D_ * D_, D_, xq16, D_, D_, D_, 1);
    k_gemm<<<dim3(SP1 / 16, D_ / 128), 256, 0, stream>>>(
        Wp16, D_, Wk16 + (size_t)h * D_ * D_, D_, xkT16, SP1, D_, D_, 2);
    k_gemm<<<dim3(SP1 / 16, (SP1 / 16 + 7) / 8), 256, 0, stream>>>(
        xq16, D_, xkT16, SP1, scores, SP1, SP1, D_, 0);
    k_softmax<<<dim3(SP1), 256, 0, stream>>>(scores, K16 + (size_t)h * SP1 * S_, A_LR + h);
  }

  // ---- L gradient-descent iterations ----
  for (int l = 0; l < L_; ++l) {
    k_cast_fk<<<g1((size_t)B_ * S_ * D_), 256, 0, stream>>>(fk32, fk16);
    k_expect<<<dim3(B_ * S_ / 16), 256, 0, stream>>>(fk16, We16, WeT16, e32, diff32, diff16);
    k_diffsum<<<g1((size_t)B_ * D_), 256, 0, stream>>>(diff32, dsum);
    k_vh<<<dim3(S_ / 32, D_ / 256, B_ * H_), 256, 0, stream>>>(diff16, Wv16, Vh16);
    k_delta<<<dim3(SP1 / 16, 2, B_), 256, 0, stream>>>(K16, Vh16, dsum, B_LR, fk32);
  }

  // ---- final logits = f_k[:, :-1] @ We^T ----
  k_cast_fk<<<g1((size_t)B_ * S_ * D_), 256, 0, stream>>>(fk32, fk16);
  k_gemm_big<<<dim3(B_ * S_ / 32, V_ / 256), 256, 0, stream>>>(
      fk16, D_, WeT16, V_, out, V_, D_, 0);
}
